// PtrDecoder_27135603376261
// MI455X (gfx1250) — matrix operand order repaired and checked by reading, compiled, not run on hardware
//
#include <hip/hip_runtime.h>
#include <math.h>

// ---------------------------------------------------------------------------
// Pointer-generator decoder step for MI455X (gfx1250, wave32, WMMA + TDM).
// Dominant cost: a = story_states @ Was^T  (65536 x 1024 x 1024 = 137 GFLOP)
// -> fused bf16 WMMA kernel; story tile staged into the 320KB WGP LDS via the
//    Tensor Data Mover (tensor_load_to_lds, TENSORcnt), then converted to bf16.
// ---------------------------------------------------------------------------

typedef __bf16 bf16;
typedef __attribute__((ext_vector_type(16))) __bf16 v16bf;
typedef __attribute__((ext_vector_type(8)))  float  v8f;

#define S_LEN 1024
#define B_SZ  64
#define H_DIM 1024
#define E_DIM 512
#define V_SZ  32000

#define LDT   36      // LDS row stride (bf16 elems) for 64x32 tiles
#define LDA_S 1032    // LDS row stride for the big 64x1024 story tile

__device__ __forceinline__ float sigmoidf_(float x) { return 1.f / (1.f + expf(-x)); }

// ------------------------- Tensor Data Mover helpers ------------------------
#if defined(__has_builtin)
#if __has_builtin(__builtin_amdgcn_tensor_load_to_lds)
#define HAVE_TDM 1
#endif
#endif

#ifdef HAVE_TDM
typedef unsigned int u32x4 __attribute__((ext_vector_type(4)));
typedef int          i32x4 __attribute__((ext_vector_type(4)));
typedef int          i32x8 __attribute__((ext_vector_type(8)));

// 2D f32 tile load, global -> LDS, per CDNA5 D# layout (ISA 08_async_tensor
// sections 8.3/8.4): group0 = {flags, lds_addr, global_addr, type=2},
// group1 = {data_size=4B, tensor dims, tile dims, dim0 stride}. Issue from a
// single wave; completion tracked with TENSORcnt.
__device__ __forceinline__ void tdm_load_2d_f32(unsigned int lds_byte_addr,
                                                const void* gptr,
                                                unsigned int rows,            // tile_dim1
                                                unsigned int row_elems,       // tile_dim0
                                                unsigned int row_stride_elems // tensor_dim0_stride
                                                ) {
  unsigned long long ga = (unsigned long long)gptr;
  u32x4 g0;
  g0[0] = 1u;                                                  // count=1, user D#
  g0[1] = lds_byte_addr;                                       // lds_addr
  g0[2] = (unsigned int)(ga & 0xFFFFFFFFu);                    // global_addr lo
  g0[3] = (unsigned int)((ga >> 32) & 0x01FFFFFFu) | (2u << 30); // addr hi | type=2
  i32x8 g1;
  g1[0] = (int)(2u << 16);                                     // data_size=2 (4 bytes)
  g1[1] = (int)((row_elems & 0xFFFFu) << 16);                  // tensor_dim0[15:0]
  g1[2] = (int)((row_elems >> 16) | ((rows & 0xFFFFu) << 16)); // dim0 hi | tensor_dim1 lo
  g1[3] = (int)((rows >> 16) | ((row_elems & 0xFFFFu) << 16)); // dim1 hi | tile_dim0
  g1[4] = (int)(rows & 0xFFFFu);                               // tile_dim1 (tile_dim2=0)
  g1[5] = (int)row_stride_elems;                               // tensor_dim0_stride lo32
  g1[6] = 0;                                                   // stride hi | dim1_stride lo
  g1[7] = 0;
  i32x4 gz = {0, 0, 0, 0};
#if __clang_major__ >= 23
  i32x8 gz8 = {0, 0, 0, 0, 0, 0, 0, 0};
  __builtin_amdgcn_tensor_load_to_lds(g0, g1, gz, gz, gz8, 0);
#else
  __builtin_amdgcn_tensor_load_to_lds(g0, g1, gz, gz, 0);
#endif
}

__device__ __forceinline__ unsigned int lds_addr_of(const void* p) {
  // flat LDS address: low 32 bits are the LDS byte offset (aperture mapping)
  return (unsigned int)(unsigned long long)p;
}
#endif  // HAVE_TDM

// One 16x16x32 bf16 WMMA from LDS-resident tiles.
// A tile: rows = M (row-major, lda elems), B tile: rows = N holding W[n][k].
// Fragment layouts per CDNA5 ISA 7.12.2 (wave32).
__device__ __forceinline__ v8f wmma_bf16_16x16x32(const bf16* lA, int lda, int mBase,
                                                  const bf16* lB, int ldb, int nBase,
                                                  v8f acc) {
  const int lane = threadIdx.x & 31;
  const int lrow = lane & 15;
  const int lhi  = lane >> 4;
  v16bf a, b;
  const bf16* ap = lA + (size_t)(mBase + lrow) * lda;
#pragma unroll
  for (int v = 0; v < 8; ++v) {
    int kb = 2 * v + ((v >= 4) ? 8 : 0) + 8 * lhi;
    a[2 * v]     = ap[kb];
    a[2 * v + 1] = ap[kb + 1];
  }
  const bf16* bp = lB + (size_t)(nBase + lrow) * ldb + 8 * lhi;
#pragma unroll
  for (int e = 0; e < 16; ++e) b[e] = bp[e + ((e >= 8) ? 8 : 0)];
  return __builtin_amdgcn_wmma_f32_16x16x32_bf16(false, a, false, b, (short)0, acc,
                                                 false, false);
}

// -------------------- generic M=64 GEMM: C = A @ W^T + bias ----------------
__global__ __launch_bounds__(256) void k_gemm64(const float* __restrict__ A, int K,
                                                const float* __restrict__ W,
                                                const float* __restrict__ bias,
                                                float* __restrict__ C, int N) {
  __shared__ bf16 lA[64 * LDT];
  __shared__ bf16 lB[64 * LDT];
  const int tid  = threadIdx.x;
  const int n0   = blockIdx.x * 64;
  const int wave = tid >> 5;
  const int lane = tid & 31;
  const int mt   = wave >> 1;
  const int ntA  = (wave & 1) * 2;
  const int row  = tid >> 2;
  const int cg   = (tid & 3) * 8;
  v8f acc0 = {0.f, 0.f, 0.f, 0.f, 0.f, 0.f, 0.f, 0.f};
  v8f acc1 = acc0;

  for (int k0 = 0; k0 < K; k0 += 32) {
    __syncthreads();
    const float4* ap4 = (const float4*)(A + (size_t)row * K + k0 + cg);
    const float4* wp4 = (const float4*)(W + (size_t)(n0 + row) * K + k0 + cg);
    float4 a0 = ap4[0], a1 = ap4[1];
    float4 w0 = wp4[0], w1 = wp4[1];
    bf16* da = lA + row * LDT + cg;
    bf16* db = lB + row * LDT + cg;
    da[0] = (bf16)a0.x; da[1] = (bf16)a0.y; da[2] = (bf16)a0.z; da[3] = (bf16)a0.w;
    da[4] = (bf16)a1.x; da[5] = (bf16)a1.y; da[6] = (bf16)a1.z; da[7] = (bf16)a1.w;
    db[0] = (bf16)w0.x; db[1] = (bf16)w0.y; db[2] = (bf16)w0.z; db[3] = (bf16)w0.w;
    db[4] = (bf16)w1.x; db[5] = (bf16)w1.y; db[6] = (bf16)w1.z; db[7] = (bf16)w1.w;
    __syncthreads();
    acc0 = wmma_bf16_16x16x32(lA, LDT, mt * 16, lB, LDT, ntA * 16, acc0);
    acc1 = wmma_bf16_16x16x32(lA, LDT, mt * 16, lB, LDT, (ntA + 1) * 16, acc1);
  }

  const int lrow = lane & 15, lhi = lane >> 4;
  {
    int n = n0 + ntA * 16 + lrow;
    float bv = bias[n];
#pragma unroll
    for (int r = 0; r < 8; ++r) {
      int m = mt * 16 + r + 8 * lhi;
      C[(size_t)m * N + n] = acc0[r] + bv;
    }
  }
  {
    int n = n0 + (ntA + 1) * 16 + lrow;
    float bv = bias[n];
#pragma unroll
    for (int r = 0; r < 8; ++r) {
      int m = mt * 16 + r + 8 * lhi;
      C[(size_t)m * N + n] = acc1[r] + bv;
    }
  }
}

// ------------------ fused attention-score kernel (dominant) -----------------
// LDS budget/block (~198KB of the 320KB WGP LDS):
//   lA: 64 x LDA_S bf16 (129KB)  | lB: 64 x LDT bf16 | covS/partS | f32 TDM buf 64KB
__global__ __launch_bounds__(256) void k_attn_scores(
    const float* __restrict__ SS,    // (S,B,H) f32
    const float* __restrict__ Was,   // (H,H)
    const float* __restrict__ qp,    // (B,H): h_new@Wah^T + (bah+bas+bac)
    const float* __restrict__ cov,   // (B,S)
    const float* __restrict__ Wac,   // (H,1)
    const float* __restrict__ Wsq,   // (1,H)
    const float* __restrict__ bsq,   // (1,)
    float* __restrict__ scores)      // (B,S)
{
  extern __shared__ char smem[];
  bf16*  lA    = (bf16*)smem;                                    // 64 x LDA_S
  bf16*  lB    = (bf16*)(smem + (size_t)64 * LDA_S * 2);         // 64 x LDT
  float* covS  = (float*)(smem + (size_t)64 * LDA_S * 2 + 64 * LDT * 2);
  float* partS = covS + 64;
  float* f32buf = partS + 64;                                    // 64 x 256 f32

  const int s   = blockIdx.x;
  const int tid = threadIdx.x;

  // ---- Stage A: story_states[s][0:64][0:1024] -> bf16 LDS tile ----
#ifdef HAVE_TDM
  // TDM: 4 chunks of 64 rows x 256 f32, async DMA into LDS, then cvt to bf16.
  for (int c = 0; c < 4; ++c) {
    if (tid < 32) {
      tdm_load_2d_f32(lds_addr_of(f32buf),
                      SS + (size_t)s * B_SZ * H_DIM + c * 256,
                      /*rows=*/64, /*row_elems=*/256, /*stride=*/H_DIM);
      __builtin_amdgcn_s_wait_tensorcnt(0);
    }
    __syncthreads();
    for (int i = tid; i < 64 * 256; i += 256) {
      int bb = i >> 8, h = i & 255;
      lA[(size_t)bb * LDA_S + c * 256 + h] = (bf16)f32buf[i];
    }
    __syncthreads();
  }
#else
  {
    const float* base = SS + (size_t)s * B_SZ * H_DIM;
    const int b  = tid >> 2;
    const int c0 = (tid & 3) * 256;
    const float4* src = (const float4*)(base + (size_t)b * H_DIM + c0);
    bf16* dst = lA + (size_t)b * LDA_S + c0;
#pragma unroll 4
    for (int j = 0; j < 64; ++j) {
      float4 v = src[j];
      dst[4 * j + 0] = (bf16)v.x; dst[4 * j + 1] = (bf16)v.y;
      dst[4 * j + 2] = (bf16)v.z; dst[4 * j + 3] = (bf16)v.w;
    }
  }
#endif
  if (tid < 64) { covS[tid] = cov[(size_t)tid * S_LEN + s]; partS[tid] = 0.f; }
  __syncthreads();

  const int wave = tid >> 5, lane = tid & 31;
  const int mt   = wave >> 1, ntA = (wave & 1) * 2;
  const int lrow = lane & 15, lhi = lane >> 4;
  const int row  = tid >> 2,  cg  = (tid & 3) * 8;

  float part[8];
#pragma unroll
  for (int r = 0; r < 8; ++r) part[r] = 0.f;

  for (int n0 = 0; n0 < H_DIM; n0 += 64) {
    v8f acc0 = {0.f, 0.f, 0.f, 0.f, 0.f, 0.f, 0.f, 0.f};
    v8f acc1 = acc0;
    for (int k0 = 0; k0 < H_DIM; k0 += 32) {
      __syncthreads();
      const float4* wp4 = (const float4*)(Was + (size_t)(n0 + row) * H_DIM + k0 + cg);
      float4 w0 = wp4[0], w1 = wp4[1];
      bf16* db = lB + row * LDT + cg;
      db[0] = (bf16)w0.x; db[1] = (bf16)w0.y; db[2] = (bf16)w0.z; db[3] = (bf16)w0.w;
      db[4] = (bf16)w1.x; db[5] = (bf16)w1.y; db[6] = (bf16)w1.z; db[7] = (bf16)w1.w;
      __syncthreads();
      acc0 = wmma_bf16_16x16x32(lA, LDA_S, mt * 16, lB, LDT, ntA * 16, acc0);
      acc1 = wmma_bf16_16x16x32(lA, LDA_S, mt * 16, lB, LDT, (ntA + 1) * 16, acc1);
    }
    // fused epilogue: tanh(a + q' + cov*Wac) dot Wsq, accumulated per b
#pragma unroll
    for (int t = 0; t < 2; ++t) {
      v8f acc = t ? acc1 : acc0;
      const int g    = n0 + (ntA + t) * 16 + lrow;
      const float wc = Wac[g];
      const float wq = Wsq[g];
#pragma unroll
      for (int r = 0; r < 8; ++r) {
        int b   = mt * 16 + r + 8 * lhi;
        float v = acc[r] + qp[(size_t)b * H_DIM + g] + covS[b] * wc;
        part[r] += tanhf(v) * wq;
      }
    }
  }
#pragma unroll
  for (int r = 0; r < 8; ++r) {
    int b = mt * 16 + r + 8 * lhi;
    atomicAdd(&partS[b], part[r]);       // ds_add_f32
  }
  __syncthreads();
  if (tid < 64) scores[(size_t)tid * S_LEN + s] = partS[tid] + bsq[0];
}

// ------------------------- small elementwise kernels ------------------------
__global__ void k_embed_concat(const int* __restrict__ dec_ids,
                               const float* __restrict__ emb,
                               const float* __restrict__ context,
                               float* __restrict__ Aic) {
  int i = blockIdx.x * 256 + threadIdx.x;
  int b = i / 1536, k = i % 1536;
  float v = (k < E_DIM) ? emb[(size_t)dec_ids[b] * E_DIM + k]
                        : context[(size_t)b * H_DIM + (k - E_DIM)];
  Aic[i] = v;
}

__global__ void k_gru(const float* __restrict__ gi, const float* __restrict__ gh,
                      const float* __restrict__ hidden, float* __restrict__ h_new) {
  int i = blockIdx.x * 256 + threadIdx.x;
  int b = i >> 10, n = i & 1023;
  const float* gib = gi + (size_t)b * 3 * H_DIM;
  const float* ghb = gh + (size_t)b * 3 * H_DIM;
  float r  = sigmoidf_(gib[n] + ghb[n]);
  float z  = sigmoidf_(gib[n + H_DIM] + ghb[n + H_DIM]);
  float nn = tanhf(gib[n + 2 * H_DIM] + r * ghb[n + 2 * H_DIM]);
  h_new[i] = (1.f - z) * nn + z * hidden[i];
}

__global__ void k_bias3(const float* a, const float* b, const float* c, float* o) {
  int i = blockIdx.x * 256 + threadIdx.x;
  o[i] = a[i] + b[i] + c[i];
}

__global__ void k_concat_hc(const float* __restrict__ h, const float* __restrict__ c,
                            float* __restrict__ hc) {
  int i = blockIdx.x * 256 + threadIdx.x;
  int b = i >> 11, k = i & 2047;
  hc[i] = (k < H_DIM) ? h[(size_t)b * H_DIM + k] : c[(size_t)b * H_DIM + k - H_DIM];
}

// ----------------------------- reductions ----------------------------------
__device__ __forceinline__ float block_max_(float v, float* red) {
  int tid = threadIdx.x;
  red[tid] = v; __syncthreads();
  for (int off = 128; off > 0; off >>= 1) {
    if (tid < off) red[tid] = fmaxf(red[tid], red[tid + off]);
    __syncthreads();
  }
  float r = red[0]; __syncthreads();
  return r;
}
__device__ __forceinline__ float block_sum_(float v, float* red) {
  int tid = threadIdx.x;
  red[tid] = v; __syncthreads();
  for (int off = 128; off > 0; off >>= 1) {
    if (tid < off) red[tid] += red[tid + off];
    __syncthreads();
  }
  float r = red[0]; __syncthreads();
  return r;
}

// softmax -> mask -> renorm -> (attn, cov_new, softmax(attn)). One block per b.
__global__ __launch_bounds__(256) void k_softmax_attn(
    const float* __restrict__ scores, const int* __restrict__ story,
    const float* __restrict__ coverage, float* __restrict__ attn_ws,
    float* __restrict__ cov_out, float* __restrict__ attn_sm_out) {
  __shared__ float red[256];
  const int b = blockIdx.x, tid = threadIdx.x;
  const float* sc = scores + (size_t)b * S_LEN;
  float v[4];
  float m = -3.4e38f;
#pragma unroll
  for (int j = 0; j < 4; ++j) { v[j] = sc[tid + 256 * j]; m = fmaxf(m, v[j]); }
  m = block_max_(m, red);
  float sum = 0.f;
#pragma unroll
  for (int j = 0; j < 4; ++j) { v[j] = expf(v[j] - m); sum += v[j]; }
  sum = block_sum_(sum, red);
  float inv = 1.f / sum, msum = 0.f;
#pragma unroll
  for (int j = 0; j < 4; ++j) {
    int s = tid + 256 * j;
    float mask = (story[(size_t)b * S_LEN + s] > 0) ? 1.f : 0.f;
    v[j] = v[j] * inv * mask;
    msum += v[j];
  }
  msum = block_sum_(msum, red);
  inv = 1.f / msum;
  float m2 = -3.4e38f;
#pragma unroll
  for (int j = 0; j < 4; ++j) { v[j] *= inv; m2 = fmaxf(m2, v[j]); }
#pragma unroll
  for (int j = 0; j < 4; ++j) {
    int s = tid + 256 * j;
    attn_ws[(size_t)b * S_LEN + s] = v[j];
    cov_out[(size_t)b * S_LEN + s] = coverage[(size_t)b * S_LEN + s] + v[j];
  }
  m2 = block_max_(m2, red);
  float s2 = 0.f, e[4];
#pragma unroll
  for (int j = 0; j < 4; ++j) { e[j] = expf(v[j] - m2); s2 += e[j]; }
  s2 = block_sum_(s2, red);
  float inv2 = 1.f / s2;
#pragma unroll
  for (int j = 0; j < 4; ++j) attn_sm_out[(size_t)b * S_LEN + tid + 256 * j] = e[j] * inv2;
}

// ctx[b][h] = sum_s attn[b][s] * SS[s][b][h].  One block per b; the 16x1024
// story slab streams through LDS via the Tensor Data Mover (TENSORcnt).
#define CTX_ROWS 16
__global__ __launch_bounds__(256) void k_ctx(const float* __restrict__ attn,
                                             const float* __restrict__ SS,
                                             float* __restrict__ ctx_out) {
  extern __shared__ float sbuf[];
  float* tile  = sbuf;                    // CTX_ROWS x 1024 f32
  float* attnS = sbuf + CTX_ROWS * H_DIM; // 1024 f32
  const int b   = blockIdx.x;
  const int tid = threadIdx.x;
  for (int i = tid; i < S_LEN; i += 256) attnS[i] = attn[(size_t)b * S_LEN + i];
  __syncthreads();
  float acc[4] = {0.f, 0.f, 0.f, 0.f};
  for (int s0 = 0; s0 < S_LEN; s0 += CTX_ROWS) {
#ifdef HAVE_TDM
    if (tid < 32) {
      tdm_load_2d_f32(lds_addr_of(tile),
                      SS + ((size_t)s0 * B_SZ + b) * H_DIM,
                      /*rows=*/CTX_ROWS, /*row_elems=*/H_DIM,
                      /*stride=*/B_SZ * H_DIM);
      __builtin_amdgcn_s_wait_tensorcnt(0);
    }
#else
    for (int i = tid; i < CTX_ROWS * H_DIM; i += 256) {
      int sr = i >> 10, h = i & 1023;
      __builtin_prefetch(SS + ((size_t)(s0 + sr + CTX_ROWS) * B_SZ + b) * H_DIM + h, 0, 0);
      tile[i] = SS[((size_t)(s0 + sr) * B_SZ + b) * H_DIM + h];
    }
#endif
    __syncthreads();
#pragma unroll
    for (int sr = 0; sr < CTX_ROWS; ++sr) {
      float w = attnS[s0 + sr];
#pragma unroll
      for (int j = 0; j < 4; ++j)
        acc[j] += w * tile[sr * H_DIM + tid + 256 * j];
    }
    __syncthreads();
  }
#pragma unroll
  for (int j = 0; j < 4; ++j)
    ctx_out[(size_t)b * H_DIM + tid + 256 * j] = acc[j];
}

__global__ __launch_bounds__(256) void k_softmax_vocab(const float* __restrict__ logits,
                                                       float* __restrict__ out) {
  __shared__ float red[256];
  const int b = blockIdx.x, tid = threadIdx.x;
  const float* lg = logits + (size_t)b * V_SZ;
  float m = -3.4e38f;
  for (int i = tid; i < V_SZ; i += 256) m = fmaxf(m, lg[i]);
  m = block_max_(m, red);
  float sum = 0.f;
  for (int i = tid; i < V_SZ; i += 256) sum += expf(lg[i] - m);
  sum = block_sum_(sum, red);
  float inv = 1.f / sum;
  float* ob = out + (size_t)b * V_SZ;
  for (int i = tid; i < V_SZ; i += 256) ob[i] = expf(lg[i] - m) * inv;
}

__global__ __launch_bounds__(256) void k_pgen(const float* __restrict__ x,
                                              const float* __restrict__ ctx,
                                              const float* __restrict__ hnew,
                                              const float* __restrict__ Wpg,
                                              const float* __restrict__ bpg,
                                              float* __restrict__ pgen) {
  __shared__ float red[256];
  const int b = blockIdx.x, tid = threadIdx.x;
  float acc = 0.f;
  for (int h = tid; h < H_DIM; h += 256)
    acc += (x[(size_t)b * H_DIM + h] + ctx[(size_t)b * H_DIM + h] +
            hnew[(size_t)b * H_DIM + h]) * Wpg[h];
  acc = block_sum_(acc, red);
  if (tid == 0) pgen[b] = sigmoidf_(acc + bpg[0]);
}

// ---------------------------------------------------------------------------
extern "C" void kernel_launch(void* const* d_in, const int* in_sizes, int n_in,
                              void* d_out, int out_size, void* d_ws, size_t ws_size,
                              hipStream_t stream) {
  const float* SS       = (const float*)d_in[0];
  const int*   story    = (const int*)  d_in[1];
  const int*   dec_ids  = (const int*)  d_in[2];
  const float* hidden   = (const float*)d_in[3];
  const float* context  = (const float*)d_in[4];
  const float* coverage = (const float*)d_in[5];
  const float* embed    = (const float*)d_in[6];
  const float* Wic = (const float*)d_in[7];   const float* bic = (const float*)d_in[8];
  const float* Wah = (const float*)d_in[9];   const float* bah = (const float*)d_in[10];
  const float* Was = (const float*)d_in[11];  const float* bas = (const float*)d_in[12];
  const float* Wac = (const float*)d_in[13];  const float* bac = (const float*)d_in[14];
  const float* Wsq = (const float*)d_in[15];  const float* bsq = (const float*)d_in[16];
  const float* Wpg = (const float*)d_in[17];  const float* bpg = (const float*)d_in[18];
  const float* Wih = (const float*)d_in[19];  const float* Whh = (const float*)d_in[20];
  const float* bih = (const float*)d_in[21];  const float* bhh = (const float*)d_in[22];
  const float* Wout = (const float*)d_in[23]; const float* bout = (const float*)d_in[24];

  // outputs (f32), concatenated: pgen, vocab, h_new, ctx, softmax(attn), cov_new
  float* out      = (float*)d_out;
  float* o_pgen   = out;
  float* o_vocab  = out + 64;
  float* o_hnew   = o_vocab + (size_t)B_SZ * V_SZ;
  float* o_ctx    = o_hnew + (size_t)B_SZ * H_DIM;
  float* o_attnsm = o_ctx + (size_t)B_SZ * H_DIM;
  float* o_cov    = o_attnsm + (size_t)B_SZ * S_LEN;

  // workspace (f32)
  float* ws      = (float*)d_ws;
  float* w_aic   = ws;
  float* w_x     = w_aic   + (size_t)64 * 1536;
  float* w_gi    = w_x     + (size_t)64 * 1024;
  float* w_gh    = w_gi    + (size_t)64 * 3072;
  float* w_q     = w_gh    + (size_t)64 * 3072;
  float* w_qbias = w_q     + (size_t)64 * 1024;
  float* w_sc    = w_qbias + 1024;
  float* w_attn  = w_sc    + (size_t)64 * 1024;
  float* w_hc    = w_attn  + (size_t)64 * 1024;
  float* w_log   = w_hc    + (size_t)64 * 2048;

  // 1) A_ic = [embed[dec_ids] | context]
  k_embed_concat<<<dim3((B_SZ * 1536) / 256), dim3(256), 0, stream>>>(
      dec_ids, embed, context, w_aic);
  // 2) x = A_ic @ Wic^T + bic
  k_gemm64<<<dim3(H_DIM / 64), dim3(256), 0, stream>>>(w_aic, 1536, Wic, bic, w_x, H_DIM);
  // 3) gi = x @ Wih^T + bih ; gh = hidden @ Whh^T + bhh
  k_gemm64<<<dim3(3 * H_DIM / 64), dim3(256), 0, stream>>>(w_x, H_DIM, Wih, bih, w_gi, 3 * H_DIM);
  k_gemm64<<<dim3(3 * H_DIM / 64), dim3(256), 0, stream>>>(hidden, H_DIM, Whh, bhh, w_gh, 3 * H_DIM);
  // 4) GRU -> h_new
  k_gru<<<dim3((B_SZ * H_DIM) / 256), dim3(256), 0, stream>>>(w_gi, w_gh, hidden, o_hnew);
  // 5) qbias = bah + bas + bac ; q' = h_new @ Wah^T + qbias
  k_bias3<<<dim3(H_DIM / 256), dim3(256), 0, stream>>>(bah, bas, bac, w_qbias);
  k_gemm64<<<dim3(H_DIM / 64), dim3(256), 0, stream>>>(o_hnew, H_DIM, Wah, w_qbias, w_q, H_DIM);
  // 6) fused attention scores (dominant WMMA kernel), ~198KB dynamic LDS
  size_t smemA = (size_t)64 * LDA_S * 2 + (size_t)64 * LDT * 2 +
                 2 * 64 * sizeof(float) + (size_t)64 * 256 * sizeof(float);
  k_attn_scores<<<dim3(S_LEN), dim3(256), smemA, stream>>>(
      SS, Was, w_q, coverage, Wac, Wsq, bsq, w_sc);
  // 7) softmax / mask / renorm -> attn, cov_new, softmax(attn)
  k_softmax_attn<<<dim3(B_SZ), dim3(256), 0, stream>>>(
      w_sc, story, coverage, w_attn, o_cov, o_attnsm);
  // 8) ctx = attn . story_states (TDM-staged LDS streaming)
  size_t smemC = (size_t)(CTX_ROWS * H_DIM + S_LEN) * sizeof(float);
  k_ctx<<<dim3(B_SZ), dim3(256), smemC, stream>>>(w_attn, SS, o_ctx);
  // 9) hc = [h_new | ctx] ; logits = hc @ Wout^T + bout
  k_concat_hc<<<dim3((B_SZ * 2 * H_DIM) / 256), dim3(256), 0, stream>>>(o_hnew, o_ctx, w_hc);
  k_gemm64<<<dim3(V_SZ / 64), dim3(256), 0, stream>>>(w_hc, 2 * H_DIM, Wout, bout, w_log, V_SZ);
  // 10) vocab softmax
  k_softmax_vocab<<<dim3(B_SZ), dim3(256), 0, stream>>>(w_log, o_vocab);
  // 11) pgen
  k_pgen<<<dim3(B_SZ), dim3(256), 0, stream>>>(w_x, o_ctx, o_hnew, Wpg, bpg, o_pgen);

  (void)in_sizes; (void)n_in; (void)out_size; (void)ws_size;
}
